// HRM_30975304139509
// MI455X (gfx1250) — matrix operand order repaired and checked by reading, not compiled, not run on hardware
//
#include <hip/hip_runtime.h>
#include <hip/hip_bf16.h>

// ---------------------------------------------------------------------------
// Types for CDNA5 WMMA (wave32): bf16 16x16x32, f32 accumulate
// ---------------------------------------------------------------------------
typedef __attribute__((ext_vector_type(16))) __bf16       v16bf;
typedef __attribute__((ext_vector_type(8)))  float        v8f;
typedef __attribute__((ext_vector_type(4)))  unsigned int v4u;

struct APair { v4u lo; v4u hi; };

__device__ __forceinline__ unsigned short f2bf(float f) {
  unsigned int u = __builtin_bit_cast(unsigned int, f);
  u += 0x7FFFu + ((u >> 16) & 1u);           // round-to-nearest-even
  return (unsigned short)(u >> 16);
}
__device__ __forceinline__ float bf2f(unsigned short h) {
  unsigned int u = ((unsigned int)h) << 16;
  return __builtin_bit_cast(float, u);
}

// A fragment, 16-bit A 16x32 layout (contiguous K, matching B):
//   lanes 0-15  : row m, K = {0..15}   (relative to kb)
//   lanes 16-31 : row m, K = {16..31}
// p points at row + kb + (hi*16); chunks at +0 and +8 halfs.
__device__ __forceinline__ v16bf load_a_frag(const unsigned short* p) {
  APair t;
  t.lo = *(const v4u*)(p);
  t.hi = *(const v4u*)(p + 8);
  return __builtin_bit_cast(v16bf, t);
}

// ---------------------------------------------------------------------------
// Generic bf16 WMMA GEMM: out[M,N] = A[M,K] @ W[N,K]^T  (W stored [N,K])
// Wave computes a 32x64 tile: 2 A fragments x 4 B fragments -> 8 accumulators.
// B fragments are reused across both A fragments (10.7 MACs/byte from L2).
// Epilogue: + bias[n], + Dvec[n]*addend[m,n] (fp32 or bf16 addend),
//           writes fp32 and/or bf16 outputs.
// Requires: M%32==0, N%64==0, K%32==0 (true for all call sites).
// ---------------------------------------------------------------------------
__global__ __launch_bounds__(256) void gemm_bf16_kernel(
    const unsigned short* __restrict__ A,
    const unsigned short* __restrict__ W,
    int M, int N, int K,
    const float* __restrict__ bias,
    const float* __restrict__ Dvec,
    const float* __restrict__ addF,
    const unsigned short* __restrict__ addB,
    float* __restrict__ outF,
    unsigned short* __restrict__ outB)
{
  const int lane = threadIdx.x & 31;
  const int wid  = threadIdx.x >> 5;
  const int tilesN4 = N >> 6;
  const int total   = (M >> 5) * tilesN4;
  const int g = blockIdx.x * 8 + wid;
  if (g >= total) return;                        // wave-uniform branch: EXEC stays all-ones
  const int tm = g / tilesN4;
  const int tn = (g - tm * tilesN4) << 6;

  const int l16 = lane & 15;
  const int hi  = lane >> 4;                     // which K-half this lane holds
  const unsigned short* A0 = A + (size_t)(tm * 32 + l16) * K + (hi << 4);
  const unsigned short* A1 = A0 + (size_t)16 * K;
  const unsigned short* W0 = W + (size_t)(tn + l16     ) * K + (hi << 4);
  const unsigned short* W1 = W + (size_t)(tn + l16 + 16) * K + (hi << 4);
  const unsigned short* W2 = W + (size_t)(tn + l16 + 32) * K + (hi << 4);
  const unsigned short* W3 = W + (size_t)(tn + l16 + 48) * K + (hi << 4);

  v8f acc[8];
  #pragma unroll
  for (int i = 0; i < 8; ++i) acc[i] = (v8f){};

  for (int kb = 0; kb < K; kb += 32) {
    v16bf a0 = load_a_frag(A0 + kb);
    v16bf a1 = load_a_frag(A1 + kb);
    v16bf b0 = *(const v16bf*)(W0 + kb);
    v16bf b1 = *(const v16bf*)(W1 + kb);
    v16bf b2 = *(const v16bf*)(W2 + kb);
    v16bf b3 = *(const v16bf*)(W3 + kb);
    acc[0] = __builtin_amdgcn_wmma_f32_16x16x32_bf16(false, a0, false, b0, (short)0, acc[0], false, false);
    acc[1] = __builtin_amdgcn_wmma_f32_16x16x32_bf16(false, a0, false, b1, (short)0, acc[1], false, false);
    acc[2] = __builtin_amdgcn_wmma_f32_16x16x32_bf16(false, a0, false, b2, (short)0, acc[2], false, false);
    acc[3] = __builtin_amdgcn_wmma_f32_16x16x32_bf16(false, a0, false, b3, (short)0, acc[3], false, false);
    acc[4] = __builtin_amdgcn_wmma_f32_16x16x32_bf16(false, a1, false, b0, (short)0, acc[4], false, false);
    acc[5] = __builtin_amdgcn_wmma_f32_16x16x32_bf16(false, a1, false, b1, (short)0, acc[5], false, false);
    acc[6] = __builtin_amdgcn_wmma_f32_16x16x32_bf16(false, a1, false, b2, (short)0, acc[6], false, false);
    acc[7] = __builtin_amdgcn_wmma_f32_16x16x32_bf16(false, a1, false, b3, (short)0, acc[7], false, false);
  }

  // C/D layout: VGPR r -> rows (r, r+8) at lanes (<16, >=16), col = lane&15.
  #pragma unroll
  for (int i = 0; i < 2; ++i) {
    const int rbase = tm * 32 + i * 16 + (hi << 3);
    #pragma unroll
    for (int t = 0; t < 4; ++t) {
      const v8f a = acc[i * 4 + t];
      const int col = tn + t * 16 + l16;
      #pragma unroll
      for (int r = 0; r < 8; ++r) {
        const int row = rbase + r;
        const size_t idx = (size_t)row * N + col;
        float v = a[r];
        if (bias) v += bias[col];
        if (Dvec) {
          float u = addF ? addF[idx] : bf2f(addB[idx]);
          v += Dvec[col] * u;
        }
        if (outF) outF[idx] = v;
        if (outB) outB[idx] = f2bf(v);
      }
    }
  }
}

// ---------------------------------------------------------------------------
// AdaLN over a D=512 row, writes bf16 into concat buffer column slice.
//   sc[b, 0:512] = scale, sc[b, 512:1024] = shift
// ---------------------------------------------------------------------------
__global__ __launch_bounds__(256) void adaln_kernel(
    const float* __restrict__ xin, const float* __restrict__ sc,
    unsigned short* __restrict__ dstB, int ldb, int offb, int Lrows)
{
  const int m = blockIdx.x;
  const int b = m / Lrows;
  const int tid = threadIdx.x;
  const float* row = xin + (size_t)m * 512;
  float v0 = row[tid], v1 = row[tid + 256];
  float s = v0 + v1, q = v0 * v0 + v1 * v1;
  #pragma unroll
  for (int off = 16; off > 0; off >>= 1) {     // wave32 reduction
    s += __shfl_xor(s, off, 32);
    q += __shfl_xor(q, off, 32);
  }
  __shared__ float ls[8], lq[8];
  const int lane = tid & 31, wid = tid >> 5;
  if (lane == 0) { ls[wid] = s; lq[wid] = q; }
  __syncthreads();
  float ts = 0.f, tq = 0.f;
  #pragma unroll
  for (int i = 0; i < 8; ++i) { ts += ls[i]; tq += lq[i]; }
  const float mu  = ts * (1.f / 512.f);
  const float var = tq * (1.f / 512.f) - mu * mu;
  const float rs  = rsqrtf(var + 1e-5f);
  const float* scb = sc + (size_t)b * 1024;
  const float o0 = (v0 - mu) * rs * (1.f + scb[tid])       + scb[512 + tid];
  const float o1 = (v1 - mu) * rs * (1.f + scb[tid + 256]) + scb[512 + tid + 256];
  unsigned short* drow = dstB + (size_t)m * ldb + offb;
  drow[tid]       = f2bf(o0);
  drow[tid + 256] = f2bf(o1);
}

// cond @ W + b   (tiny: 8x1024 outputs, K=512)
__global__ void cond_gemm_kernel(const float* __restrict__ t, const float* __restrict__ W,
                                 const float* __restrict__ bias, float* __restrict__ out,
                                 int Bn, int K, int N)
{
  int i = blockIdx.x * blockDim.x + threadIdx.x;
  if (i >= Bn * N) return;
  int b = i / N, n = i - b * N;
  float acc = bias[n];
  for (int k = 0; k < K; ++k) acc += t[(size_t)b * K + k] * W[(size_t)k * N + n];
  out[i] = acc;
}

// ZOH discretization: Abar = exp(-exp(logA)*exp(logdt)); Bbar = ((Abar-1)/A)*Bm (bf16)
__global__ void discretize_kernel(const float* __restrict__ logA, const float* __restrict__ logdt,
                                  const float* __restrict__ Bm, int P, int H,
                                  float* __restrict__ Abar, unsigned short* __restrict__ Bbar)
{
  int p = threadIdx.x;
  if (p >= P) return;
  float A  = -expf(logA[p]);
  float dt = expf(logdt[p]);
  float ab = expf(A * dt);
  Abar[p] = ab;
  float coef = (ab - 1.f) / A;
  const float* srow = Bm + (size_t)p * H;
  unsigned short* drow = Bbar + (size_t)p * H;
  for (int h = 0; h < H; ++h) drow[h] = f2bf(coef * srow[h]);
}

// flat fp32 -> bf16
__global__ void convert_kernel(const float* __restrict__ src, unsigned short* __restrict__ dst, long n)
{
  long i = (long)blockIdx.x * blockDim.x + threadIdx.x;
  if (i < n) dst[i] = f2bf(src[i]);
}

// src[K,N] fp32 -> dst[N,K] bf16 (weight transpose for WMMA B operand)
__global__ void tconv_kernel(const float* __restrict__ src, unsigned short* __restrict__ dst, int K, int N)
{
  long i = (long)blockIdx.x * blockDim.x + threadIdx.x;
  if (i >= (long)K * N) return;
  long k = i / N; int n = (int)(i - k * N);
  dst[(size_t)n * K + k] = f2bf(src[i]);
}

// x[M,D] fp32 -> concat buffer bf16 at column offset
__global__ void fillx_kernel(const float* __restrict__ x, unsigned short* __restrict__ dst,
                             int Dd, int ldb, int offb, long Mrows)
{
  long i = (long)blockIdx.x * blockDim.x + threadIdx.x;
  if (i >= Mrows * Dd) return;
  long m = i / Dd; int d = (int)(i - m * Dd);
  dst[m * (size_t)ldb + offb + d] = f2bf(x[i]);
}

// diagonal SSM recurrence x_l = Abar*x_{l-1} + Bu_l, one thread per (b,p) channel
__global__ void scan_kernel(const float* __restrict__ Bu, const float* __restrict__ Abar,
                            unsigned short* __restrict__ xsb, int Lr, int P)
{
  int b = blockIdx.x, p = threadIdx.x;
  float a = Abar[p], xv = 0.f;
  const float* src = Bu + (size_t)b * Lr * P + p;
  unsigned short* dst = xsb + (size_t)b * Lr * P + p;
  for (int l = 0; l < Lr; ++l) {
    xv = a * xv + src[(size_t)l * P];
    dst[(size_t)l * P] = f2bf(xv);
  }
}

// ---------------------------------------------------------------------------
extern "C" void kernel_launch(void* const* d_in, const int* in_sizes, int n_in,
                              void* d_out, int out_size, void* d_ws, size_t ws_size,
                              hipStream_t stream)
{
  (void)in_sizes; (void)n_in; (void)out_size; (void)ws_size;
  constexpr int B = 8, L = 1024, D = 512, M = B * L;
  constexpr int Pf = 64, Ps = 256, KF = 3 * D, KS = 2 * D;

  const float* zH        = (const float*)d_in[0];
  const float* zL        = (const float*)d_in[1];
  const float* x         = (const float*)d_in[2];
  const float* tvec      = (const float*)d_in[3];
  const float* f_normH_W = (const float*)d_in[4];
  const float* f_normH_b = (const float*)d_in[5];
  const float* f_normL_W = (const float*)d_in[6];
  const float* f_normL_b = (const float*)d_in[7];
  const float* f_inj_W   = (const float*)d_in[8];
  const float* f_inj_b   = (const float*)d_in[9];
  const float* f_logA    = (const float*)d_in[10];
  const float* f_logdt   = (const float*)d_in[11];
  const float* f_B       = (const float*)d_in[12];
  const float* f_C       = (const float*)d_in[13];
  const float* f_D       = (const float*)d_in[14];
  const float* s_normH_W = (const float*)d_in[15];
  const float* s_normH_b = (const float*)d_in[16];
  const float* s_normL_W = (const float*)d_in[17];
  const float* s_normL_b = (const float*)d_in[18];
  const float* s_logA    = (const float*)d_in[19];
  const float* s_logdt   = (const float*)d_in[20];
  const float* s_B       = (const float*)d_in[21];
  const float* s_C       = (const float*)d_in[22];
  const float* s_D       = (const float*)d_in[23];
  const float* s_proj_W  = (const float*)d_in[24];
  const float* s_proj_b  = (const float*)d_in[25];

  char* ws = (char*)d_ws;
  size_t off = 0;
  auto take = [&](size_t bytes) -> char* {
    char* p = ws + off;
    off = (off + bytes + 255) & ~(size_t)255;
    return p;
  };

  // --- persistent region ---------------------------------------------------
  float* zHs  = (float*)take((size_t)M * D * 4);
  float* zLs  = (float*)take((size_t)M * D * 4);
  float* scfH = (float*)take((size_t)B * 2 * D * 4);
  float* scfL = (float*)take((size_t)B * 2 * D * 4);
  float* scsH = (float*)take((size_t)B * 2 * D * 4);
  float* scsL = (float*)take((size_t)B * 2 * D * 4);
  float* fAbar = (float*)take(Pf * 4);
  float* sAbar = (float*)take(Ps * 4);
  unsigned short* fBbar   = (unsigned short*)take((size_t)Pf * D * 2);   // [Pf, D]
  unsigned short* fCb     = (unsigned short*)take((size_t)D * Pf * 2);   // [D, Pf]
  unsigned short* sBbar   = (unsigned short*)take((size_t)Ps * KS * 2);  // [Ps, 2D]
  unsigned short* sCb     = (unsigned short*)take((size_t)KS * Ps * 2);  // [2D, Ps]
  unsigned short* finjWt  = (unsigned short*)take((size_t)D * KF * 2);   // [D, 3D]
  unsigned short* sprojWt = (unsigned short*)take((size_t)D * KS * 2);   // [D, 2D]

  // --- union scratch: fast and slow phases alias the same region -----------
  const size_t ubase = off;
  unsigned short* hcatF = (unsigned short*)take((size_t)M * KF * 2);
  float*          hF    = (float*)take((size_t)M * D * 4);
  unsigned short* hFb   = (unsigned short*)take((size_t)M * D * 2);
  float*          BuF   = (float*)take((size_t)M * Pf * 4);
  unsigned short* xsFb  = (unsigned short*)take((size_t)M * Pf * 2);
  off = ubase;
  unsigned short* hcatS = (unsigned short*)take((size_t)M * KS * 2);
  float*          BuS   = (float*)take((size_t)M * Ps * 4);
  unsigned short* xsSb  = (unsigned short*)take((size_t)M * Ps * 2);
  unsigned short* ySb   = (unsigned short*)take((size_t)M * KS * 2);

  auto gemm = [&](const unsigned short* Ain, const unsigned short* Win, int m, int n, int k,
                  const float* bias, const float* Dv, const float* aF, const unsigned short* aB,
                  float* oF, unsigned short* oB) {
    int tiles = (m >> 5) * (n >> 6);
    int grid  = (tiles + 7) / 8;
    gemm_bf16_kernel<<<grid, 256, 0, stream>>>(Ain, Win, m, n, k, bias, Dv, aF, aB, oF, oB);
  };

  // --- one-time setup ------------------------------------------------------
  hipMemcpyAsync(zHs, zH, (size_t)M * D * 4, hipMemcpyDeviceToDevice, stream);
  hipMemcpyAsync(zLs, zL, (size_t)M * D * 4, hipMemcpyDeviceToDevice, stream);

  const int cg = (B * 2 * D + 255) / 256;
  cond_gemm_kernel<<<cg, 256, 0, stream>>>(tvec, f_normH_W, f_normH_b, scfH, B, D, 2 * D);
  cond_gemm_kernel<<<cg, 256, 0, stream>>>(tvec, f_normL_W, f_normL_b, scfL, B, D, 2 * D);
  cond_gemm_kernel<<<cg, 256, 0, stream>>>(tvec, s_normH_W, s_normH_b, scsH, B, D, 2 * D);
  cond_gemm_kernel<<<cg, 256, 0, stream>>>(tvec, s_normL_W, s_normL_b, scsL, B, D, 2 * D);

  discretize_kernel<<<1, Pf, 0, stream>>>(f_logA, f_logdt, f_B, Pf, D,  fAbar, fBbar);
  discretize_kernel<<<1, Ps, 0, stream>>>(s_logA, s_logdt, s_B, Ps, KS, sAbar, sBbar);

  convert_kernel<<<(int)(((long)D * Pf + 255) / 256), 256, 0, stream>>>(f_C, fCb, (long)D * Pf);
  convert_kernel<<<(int)(((long)KS * Ps + 255) / 256), 256, 0, stream>>>(s_C, sCb, (long)KS * Ps);
  tconv_kernel<<<(int)(((long)KF * D + 255) / 256), 256, 0, stream>>>(f_inj_W,  finjWt,  KF, D);
  tconv_kernel<<<(int)(((long)KS * D + 255) / 256), 256, 0, stream>>>(s_proj_W, sprojWt, KS, D);

  // --- layer bodies --------------------------------------------------------
  auto run_fast = [&]() {
    adaln_kernel<<<M, 256, 0, stream>>>(zHs, scfH, hcatF, KF, 0, L);
    adaln_kernel<<<M, 256, 0, stream>>>(zLs, scfL, hcatF, KF, D, L);
    fillx_kernel<<<(int)(((long)M * D + 255) / 256), 256, 0, stream>>>(x, hcatF, D, KF, 2 * D, (long)M);
    // h = concat @ f_inj_W + b           (fp32 + bf16 copies)
    gemm(hcatF, finjWt, M, D, KF, f_inj_b, nullptr, nullptr, nullptr, hF, hFb);
    // Bu = h @ Bbar^T
    gemm(hFb, fBbar, M, Pf, D, nullptr, nullptr, nullptr, nullptr, BuF, nullptr);
    scan_kernel<<<B, Pf, 0, stream>>>(BuF, fAbar, xsFb, L, Pf);
    // zL = xs @ C^T + D*h
    gemm(xsFb, fCb, M, D, Pf, nullptr, f_D, hF, nullptr, zLs, nullptr);
  };
  auto run_slow = [&]() {
    adaln_kernel<<<M, 256, 0, stream>>>(zHs, scsH, hcatS, KS, 0, L);
    adaln_kernel<<<M, 256, 0, stream>>>(zLs, scsL, hcatS, KS, D, L);
    // Bu = h @ Bbar^T
    gemm(hcatS, sBbar, M, Ps, KS, nullptr, nullptr, nullptr, nullptr, BuS, nullptr);
    scan_kernel<<<B, Ps, 0, stream>>>(BuS, sAbar, xsSb, L, Ps);
    // y = xs @ C^T + D*h                 (bf16 out, fused)
    gemm(xsSb, sCb, M, KS, Ps, nullptr, s_D, nullptr, hcatS, nullptr, ySb);
    // zH = y @ proj_W + b
    gemm(ySb, sprojWt, M, D, KS, s_proj_b, nullptr, nullptr, nullptr, zHs, nullptr);
  };

  // reference schedule: F, F, S, F, F, S
  run_fast();
  run_fast();
  run_slow();
  run_fast();
  run_fast();
  run_slow();

  // outputs: (zH, zL) concatenated
  hipMemcpyAsync(d_out, zHs, (size_t)M * D * 4, hipMemcpyDeviceToDevice, stream);
  hipMemcpyAsync((char*)d_out + (size_t)M * D * 4, zLs, (size_t)M * D * 4,
                 hipMemcpyDeviceToDevice, stream);
}
